// BasicAttention_38534446580153
// MI455X (gfx1250) — compile-verified, matrix operand order checked by reading, not run on hardware
//
#include <hip/hip_runtime.h>
#include <hip/hip_fp16.h>

typedef __attribute__((ext_vector_type(16))) _Float16 v16h;
typedef __attribute__((ext_vector_type(8)))  _Float16 v8h;
typedef __attribute__((ext_vector_type(8)))  float    v8f;

#define BSZ 4
#define SEQ 2048
#define EMB 1024
#define DIM 1024
#define ROWS (BSZ * SEQ)  // 8192

static __device__ __forceinline__ v16h cat8(v8h lo, v8h hi) {
  v16h r;
#pragma unroll
  for (int i = 0; i < 8; ++i) { r[i] = lo[i]; r[i + 8] = hi[i]; }
  return r;
}

// ---------------------------------------------------------------------------
// Kernel 0: transpose + f32->f16 convert the three weight matrices.
// Wt[d][e] = (f16) W[e][d]  so that B-fragment (column) reads become
// contiguous row reads for the WMMA GEMMs.
// ---------------------------------------------------------------------------
__global__ __launch_bounds__(256) void wtrans_kernel(
    const float* __restrict__ Wq, const float* __restrict__ Wk,
    const float* __restrict__ Wv, _Float16* __restrict__ wt) {
  const float* W = (blockIdx.z == 0) ? Wq : (blockIdx.z == 1) ? Wk : Wv;
  _Float16* o = wt + (size_t)blockIdx.z * EMB * DIM;
  int idx = blockIdx.x * blockDim.x + threadIdx.x;  // [0, 1M)
  int d = idx >> 10, e = idx & 1023;
  o[(size_t)d * EMB + e] = (_Float16)W[(size_t)e * DIM + d];
}

// ---------------------------------------------------------------------------
// Kernel 1: fused Q/K/V projection GEMM via WMMA f32_16x16x32_f16.
// grid = (ROWS/16, DIM/256, 3).  Block = 256 threads = 8 waves.
// The block stages its 16x1024 x-panel into LDS as f16 ONCE (cooperative
// load + convert), so the K-loop has zero f32->f16 conversions and the
// A fragment comes from two ds_load_b128.  Each wave computes TWO 16x16
// output tiles, reusing the A fragment across both WMMAs.
// z==0 -> qh[s][d], z==1 -> kh[s][d], z==2 -> vt[d][s] (transposed store).
// ---------------------------------------------------------------------------
__global__ __launch_bounds__(256) void qkv_proj_kernel(
    const float* __restrict__ x, const _Float16* __restrict__ wt_all,
    const float* __restrict__ bq, const float* __restrict__ bk,
    const float* __restrict__ bv, _Float16* __restrict__ qh,
    _Float16* __restrict__ kh, _Float16* __restrict__ vt) {
  const int z = blockIdx.z;
  const float* bias = (z == 0) ? bq : (z == 1) ? bk : bv;
  const _Float16* wt = wt_all + (size_t)z * EMB * DIM;

  const int tid = threadIdx.x, wave = tid >> 5, lane = tid & 31;
  const int rt = blockIdx.x;  // 16-row tile
  const int l15 = lane & 15, hi = lane >> 4;
  const int koff = hi * 8;

  __shared__ _Float16 s_xa[16][EMB];  // 32 KB: f16 copy of this row panel

  // ---- Cooperative stage: 16x1024 f32 -> f16 in LDS ----
  {
    int row = tid >> 4;          // 0..15
    int seg = (tid & 15) * 64;   // 64 columns per thread
    const float* src = x + (size_t)(rt * 16 + row) * EMB + seg;
    _Float16* dstp = &s_xa[row][seg];
#pragma unroll
    for (int i = 0; i < 64; i += 8) {
      float4 f0 = *(const float4*)(src + i);
      float4 f1 = *(const float4*)(src + i + 4);
      v8h h;
      h[0] = (_Float16)f0.x; h[1] = (_Float16)f0.y;
      h[2] = (_Float16)f0.z; h[3] = (_Float16)f0.w;
      h[4] = (_Float16)f1.x; h[5] = (_Float16)f1.y;
      h[6] = (_Float16)f1.z; h[7] = (_Float16)f1.w;
      *(v8h*)(dstp + i) = h;
    }
  }
  __syncthreads();

  // ---- K-loop: 2 output tiles per wave, A fragment reused ----
  const int n0 = blockIdx.y * 256 + wave * 32 + l15;  // tile 0 column
  const int n1 = n0 + 16;                             // tile 1 column
  const _Float16* w0 = wt + (size_t)n0 * EMB;
  const _Float16* w1 = wt + (size_t)n1 * EMB;

  v8f acc0 = {}, acc1 = {};
#pragma unroll 4
  for (int e0 = 0; e0 < EMB; e0 += 32) {
    int base = e0 + koff;
    __builtin_prefetch(w0 + base + 256, 0, 1);
    __builtin_prefetch(w1 + base + 256, 0, 1);
    v16h a  = cat8(*(const v8h*)&s_xa[l15][base],
                   *(const v8h*)&s_xa[l15][base + 16]);
    v16h b0 = cat8(*(const v8h*)(w0 + base), *(const v8h*)(w0 + base + 16));
    v16h b1 = cat8(*(const v8h*)(w1 + base), *(const v8h*)(w1 + base + 16));
    acc0 = __builtin_amdgcn_wmma_f32_16x16x32_f16(false, a, false, b0,
                                                  (short)0, acc0, false, false);
    acc1 = __builtin_amdgcn_wmma_f32_16x16x32_f16(false, a, false, b1,
                                                  (short)0, acc1, false, false);
  }

  float bv0 = bias[n0], bv1 = bias[n1];
  if (z < 2) {
    _Float16* dst = (z == 0) ? qh : kh;
#pragma unroll
    for (int r = 0; r < 8; ++r) {
      int gr = rt * 16 + r + hi * 8;
      dst[(size_t)gr * DIM + n0] = (_Float16)(acc0[r] + bv0);
      dst[(size_t)gr * DIM + n1] = (_Float16)(acc1[r] + bv1);
    }
  } else {
#pragma unroll
    for (int r = 0; r < 8; ++r) {
      int gr = rt * 16 + r + hi * 8;
      vt[(size_t)n0 * ROWS + gr] = (_Float16)(acc0[r] + bv0);
      vt[(size_t)n1 * ROWS + gr] = (_Float16)(acc1[r] + bv1);
    }
  }
}

// ---------------------------------------------------------------------------
// Kernel 2: flash-style attention.  One block per (batch, 16-row query tile).
// 8 waves: per 128-key chunk each wave computes one 16x16 score tile (QK^T
// over E via WMMA); the online softmax is parallelized across all 256
// threads (each owns 8 score columns of one row, LDS tree-reduced); then
// all waves accumulate their 128-wide D-slice with P@V WMMAs.
// ---------------------------------------------------------------------------
__global__ __launch_bounds__(256) void attn_kernel(
    const _Float16* __restrict__ qh, const _Float16* __restrict__ kh,
    const _Float16* __restrict__ vt, const int* __restrict__ mask,
    float* __restrict__ out) {
  const int tid = threadIdx.x, wave = tid >> 5, lane = tid & 31;
  const int b = blockIdx.x >> 7;
  const int qbase = (blockIdx.x & 127) * 16;
  const int l15 = lane & 15, hi = lane >> 4;
  const int koff = hi * 8;

  __shared__ float    s_scores[16][128];
  __shared__ _Float16 s_probs[16][128];
  __shared__ float    s_red[16][16];
  __shared__ float    s_m[16], s_l[16], s_alpha[16];

  if (tid < 16) { s_m[tid] = -1e30f; s_l[tid] = 0.0f; }
  __syncthreads();

  const v8f vzero = {};
  v8f acc[8];
#pragma unroll
  for (int t = 0; t < 8; ++t) acc[t] = vzero;

  const int dbase = wave * 128;
  const int srow_sm = tid >> 4;        // softmax: row owned by this thread
  const int scol_sm = (tid & 15) * 8;  // softmax: first of 8 owned columns

  for (int kc = 0; kc < 16; ++kc) {
    const int kbase = kc * 128;

    // ---- Phase 1: this wave's 16-key score tile over full E ----
    {
      const _Float16* qrow = qh + (size_t)(b * SEQ + qbase + l15) * EMB;
      const _Float16* krow = kh + (size_t)(b * SEQ + kbase + wave * 16 + l15) * EMB;
      v8f sacc = vzero;
#pragma unroll 4
      for (int e0 = 0; e0 < EMB; e0 += 32) {
        int base = e0 + koff;
        v16h a  = cat8(*(const v8h*)(qrow + base), *(const v8h*)(qrow + base + 16));
        v16h bb = cat8(*(const v8h*)(krow + base), *(const v8h*)(krow + base + 16));
        sacc = __builtin_amdgcn_wmma_f32_16x16x32_f16(false, a, false, bb,
                                                      (short)0, sacc, false, false);
      }
#pragma unroll
      for (int r = 0; r < 8; ++r) {
        int row = r + hi * 8;
        int col = wave * 16 + l15;
        int mv = mask[(size_t)(b * SEQ + qbase + row) * SEQ + kbase + col];
        float sc = sacc[r] * (1.0f / 1024.0f);  // scale by QD, not sqrt(QD)
        s_scores[row][col] = (mv == 0) ? -1e30f : sc;
      }
    }
    __syncthreads();

    // ---- Phase 2: online softmax, parallel over all 256 threads ----
    {
      float lm = -1e30f;
#pragma unroll
      for (int j = 0; j < 8; ++j) lm = fmaxf(lm, s_scores[srow_sm][scol_sm + j]);
      s_red[srow_sm][tid & 15] = lm;
    }
    __syncthreads();
    if (tid < 16) {
      float cmax = -1e30f;
#pragma unroll
      for (int j = 0; j < 16; ++j) cmax = fmaxf(cmax, s_red[tid][j]);
      float mold = s_m[tid];
      float mnew = fmaxf(mold, cmax);
      s_alpha[tid] = __expf(mold - mnew);
      s_m[tid] = mnew;
    }
    __syncthreads();
    {
      float mnew = s_m[srow_sm];
      float sum = 0.0f;
#pragma unroll
      for (int j = 0; j < 8; ++j) {
        float p = __expf(s_scores[srow_sm][scol_sm + j] - mnew);
        s_probs[srow_sm][scol_sm + j] = (_Float16)p;
        sum += p;
      }
      s_red[srow_sm][tid & 15] = sum;
    }
    __syncthreads();
    if (tid < 16) {
      float sum = 0.0f;
#pragma unroll
      for (int j = 0; j < 16; ++j) sum += s_red[tid][j];
      s_l[tid] = s_l[tid] * s_alpha[tid] + sum;
    }
    __syncthreads();

    // ---- Phase 3: rescale accumulators, accumulate P @ V ----
    float al[8];
#pragma unroll
    for (int r = 0; r < 8; ++r) al[r] = s_alpha[r + hi * 8];
#pragma unroll
    for (int t = 0; t < 8; ++t)
#pragma unroll
      for (int r = 0; r < 8; ++r) acc[t][r] *= al[r];

#pragma unroll
    for (int ks = 0; ks < 4; ++ks) {
      int pb = ks * 32 + koff;
      v16h pa = cat8(*(const v8h*)&s_probs[l15][pb],
                     *(const v8h*)&s_probs[l15][pb + 16]);
      int srow = b * SEQ + kbase + ks * 32 + koff;
#pragma unroll
      for (int t = 0; t < 8; ++t) {
        const _Float16* vrow = vt + (size_t)(dbase + t * 16 + l15) * ROWS + srow;
        v16h vb = cat8(*(const v8h*)(vrow), *(const v8h*)(vrow + 16));
        acc[t] = __builtin_amdgcn_wmma_f32_16x16x32_f16(false, pa, false, vb,
                                                        (short)0, acc[t], false, false);
      }
    }
    __syncthreads();
  }

  // ---- Finalize: divide by running denominator, store f32 output ----
  float li[8];
#pragma unroll
  for (int r = 0; r < 8; ++r) li[r] = 1.0f / s_l[r + hi * 8];
#pragma unroll
  for (int t = 0; t < 8; ++t)
#pragma unroll
    for (int r = 0; r < 8; ++r) {
      int row = qbase + r + hi * 8;
      out[(size_t)(b * SEQ + row) * DIM + dbase + t * 16 + l15] = acc[t][r] * li[r];
    }
}

// ---------------------------------------------------------------------------
// Launch: inputs are x, mask, Wq, bq, Wk, bk, Wv, bv.
// Workspace layout (f16): qh[8192][1024] | kh[8192][1024] | vt[1024][8192]
//                         | Wt (3 x [1024][1024])  -> 54 MB total.
// ---------------------------------------------------------------------------
extern "C" void kernel_launch(void* const* d_in, const int* in_sizes, int n_in,
                              void* d_out, int out_size, void* d_ws, size_t ws_size,
                              hipStream_t stream) {
  const float* x    = (const float*)d_in[0];
  const int*   mask = (const int*)d_in[1];
  const float* Wq   = (const float*)d_in[2];
  const float* bq   = (const float*)d_in[3];
  const float* Wk   = (const float*)d_in[4];
  const float* bk   = (const float*)d_in[5];
  const float* Wv   = (const float*)d_in[6];
  const float* bv   = (const float*)d_in[7];
  float* out = (float*)d_out;

  _Float16* qh = (_Float16*)d_ws;
  _Float16* kh = qh + (size_t)ROWS * DIM;
  _Float16* vt = kh + (size_t)ROWS * DIM;
  _Float16* wt = vt + (size_t)ROWS * DIM;

  dim3 bt(256);
  wtrans_kernel<<<dim3(EMB * DIM / 256, 1, 3), bt, 0, stream>>>(Wq, Wk, Wv, wt);
  qkv_proj_kernel<<<dim3(ROWS / 16, DIM / 256, 3), bt, 0, stream>>>(
      x, wt, bq, bk, bv, qh, kh, vt);
  attn_kernel<<<dim3(BSZ * (SEQ / 16)), bt, 0, stream>>>(qh, kh, vt, mask, out);
}
